// MultiViewGAT_62311385531065
// MI455X (gfx1250) — matrix operand order repaired and checked by reading, not compiled, not run on hardware
//
#include <hip/hip_runtime.h>
#include <hip/hip_bf16.h>

typedef __attribute__((ext_vector_type(16))) __bf16 v16bf;
typedef __attribute__((ext_vector_type(8)))  __bf16 v8bf;
typedef __attribute__((ext_vector_type(8)))  float  v8f;
typedef __attribute__((ext_vector_type(4)))  int    v4i;

#define N 8192
#define F 128
#define ALPHA 0.2f
#define LOG2E 1.44269504088896f

// Adjacency LDS tile: 16 rows x 256 bytes, rows padded to 272B (68 dwords)
// so ds_load_b64 reads from all 32 lanes are bank-conflict-free.
#define AROW 272

#if defined(__has_builtin)
#if __has_builtin(__builtin_amdgcn_global_load_async_to_lds_b128)
#define HAS_ASYNC_LDS 1
#endif
#if __has_builtin(__builtin_amdgcn_s_wait_asynccnt)
#define HAS_WAIT_ASYNC 1
#endif
#endif

typedef __attribute__((address_space(1))) v4i* gv4i_p;
typedef __attribute__((address_space(3))) v4i* lv4i_p;

__device__ __forceinline__ float leaky(float v) { return v > 0.f ? v : ALPHA * v; }

// issue one 16B global->LDS copy (async if available, sync fallback otherwise)
__device__ __forceinline__ void copy16_to_lds(const unsigned char* g, unsigned char* l) {
#ifdef HAS_ASYNC_LDS
  __builtin_amdgcn_global_load_async_to_lds_b128((gv4i_p)g, (lv4i_p)l, 0, 0);
#else
  *(uint4*)l = *(const uint4*)g;
#endif
}

__device__ __forceinline__ void wait_async_le1() {
#ifdef HAS_ASYNC_LDS
#ifdef HAS_WAIT_ASYNC
  __builtin_amdgcn_s_wait_asynccnt(1);
#else
  asm volatile("s_wait_asynccnt 1" ::: "memory");
#endif
#endif
}
__device__ __forceinline__ void wait_async_le0() {
#ifdef HAS_ASYNC_LDS
#ifdef HAS_WAIT_ASYNC
  __builtin_amdgcn_s_wait_asynccnt(0);
#else
  asm volatile("s_wait_asynccnt 0" ::: "memory");
#endif
#endif
}

// ---------------------------------------------------------------------------
// Kernel 1: Wh = X @ W  (bf16 WMMA, f32 accum), store transposed bf16 WhT[F][N]
// grid (N/16, 2 layers), block 256 (8 waves, wave w -> 16-col tile w)
// ---------------------------------------------------------------------------
__global__ __launch_bounds__(256) void k_gemm_wh(
    const float* __restrict__ xall,   // [2][N][F]
    const float* __restrict__ W0,     // [F][F]
    const float* __restrict__ W1,
    __bf16* __restrict__ whTall)      // [2][F][N]
{
  const int layer = blockIdx.y;
  const int r0   = blockIdx.x * 16;
  const int lane = threadIdx.x & 31;
  const int w    = threadIdx.x >> 5;
  const int li   = lane & 15;
  const int hi   = lane >> 4;

  const float* x = xall + (size_t)layer * N * F;
  const float* W = layer ? W1 : W0;
  __bf16* wt     = whTall + (size_t)layer * F * N;

  const int n   = 16 * w + li;   // output column (B lane = N)
  const int row = r0 + li;       // A lane = M

  v8f acc = {};
  for (int kt = 0; kt < 4; ++kt) {
    const int kb = kt * 32;
    // A 16x32 bf16: lane li=M; elems 0..7 -> K = kb+8*hi+e, 8..15 -> K = kb+16+8*hi+e
    const float* xr = x + (size_t)row * F + kb + 8 * hi;
    v16bf va;
#pragma unroll
    for (int e = 0; e < 8; ++e) va[e]     = (__bf16)xr[e];
#pragma unroll
    for (int e = 0; e < 8; ++e) va[8 + e] = (__bf16)xr[16 + e];
    // B 32x16 bf16: lane li=N; elems 0..7 -> K = kb+8*hi+e, 8..15 -> K = kb+16+8*hi+e
    v16bf vb;
#pragma unroll
    for (int e = 0; e < 16; ++e)
      vb[e] = (__bf16)W[(size_t)(kb + 8 * hi + (e < 8 ? e : 8 + e)) * F + n];
    acc = __builtin_amdgcn_wmma_f32_16x16x32_bf16(false, va, false, vb,
                                                  (short)0, acc, false, false);
  }
  // C layout: VGPR v -> row r0 + v + 8*hi, col n.  Store transposed (contiguous rows)
  v8bf vo;
#pragma unroll
  for (int v = 0; v < 8; ++v) vo[v] = (__bf16)acc[v];
  *(v8bf*)(wt + (size_t)n * N + r0 + 8 * hi) = vo;
}

// ---------------------------------------------------------------------------
// Kernel 2: f1_i = Wh[i,:] . a[:F], f2_i = Wh[i,:] . a[F:]
// ---------------------------------------------------------------------------
__global__ __launch_bounds__(256) void k_f1f2(
    const __bf16* __restrict__ whTall,
    const float* __restrict__ a0, const float* __restrict__ a1,
    float* __restrict__ f1, float* __restrict__ f2)
{
  const int layer = blockIdx.y;
  const int i = blockIdx.x * blockDim.x + threadIdx.x;
  const __bf16* wt = whTall + (size_t)layer * F * N;
  const float* a = layer ? a1 : a0;
  float s1 = 0.f, s2 = 0.f;
#pragma unroll 4
  for (int c = 0; c < F; ++c) {
    float v = (float)wt[(size_t)c * N + i];
    s1 += v * a[c];
    s2 += v * a[F + c];
  }
  f1[layer * N + i] = s1;
  f2[layer * N + i] = s2;
}

// ---------------------------------------------------------------------------
// Kernel 3: rowmax_i = leakyrelu(f1_i + max_{j in adj(i)} f2_j)   (monotonicity)
// grid (N, 2), block 256
// ---------------------------------------------------------------------------
__global__ __launch_bounds__(256) void k_rowmax(
    const unsigned char* __restrict__ adjall,
    const float* __restrict__ f1, const float* __restrict__ f2,
    float* __restrict__ rowmax)
{
  const int layer = blockIdx.y;
  const int i = blockIdx.x;
  const unsigned char* adj = adjall + (size_t)layer * N * N + (size_t)i * N;
  const float* f2l = f2 + layer * N;
  float m = -3.0e38f;
  for (int j0 = threadIdx.x * 16; j0 < N; j0 += 256 * 16) {
    uint4 av = *(const uint4*)(adj + j0);
    const unsigned char* b = (const unsigned char*)&av;
#pragma unroll
    for (int t = 0; t < 16; ++t)
      if (b[t]) m = fmaxf(m, f2l[j0 + t]);
  }
  __shared__ float red[256];
  red[threadIdx.x] = m;
  __syncthreads();
  for (int s = 128; s > 0; s >>= 1) {
    if (threadIdx.x < s) red[threadIdx.x] = fmaxf(red[threadIdx.x], red[threadIdx.x + s]);
    __syncthreads();
  }
  if (threadIdx.x == 0) {
    float mm = red[0];
    rowmax[layer * N + i] = (mm < -1.0e37f) ? 0.f : leaky(f1[layer * N + i] + mm);
  }
}

// ---------------------------------------------------------------------------
// Kernel 4: fused masked-softmax attention:  h = softmax_row(mask(e)) @ Wh
// Flash-style: adjacency tiles double-buffered in LDS via async global->LDS
// copies (ASYNCcnt pipelined against the WMMAs); P tiles built in registers
// (8 waves split the exp work), staged via LDS in WMMA A-layout, consumed by
// 8 bf16 WMMAs per wave.  grid (N/16, 2), block 256
// ---------------------------------------------------------------------------
__global__ __launch_bounds__(256) void k_attn(
    const unsigned char* __restrict__ adjall,
    const __bf16* __restrict__ whTall,
    const float* __restrict__ f1all, const float* __restrict__ f2all,
    const float* __restrict__ rowmaxall,
    float* __restrict__ hall)         // [2][N][F]
{
  const int layer = blockIdx.y;
  const int r0   = blockIdx.x * 16;
  const int lane = threadIdx.x & 31;
  const int w    = threadIdx.x >> 5;
  const int li   = lane & 15;
  const int hi   = lane >> 4;

  const __bf16* whT = whTall + (size_t)layer * F * N;
  const float* f1   = f1all + layer * N;
  const float* f2   = f2all + layer * N;
  const float* rmax = rowmaxall + layer * N;
  float* h          = hall + (size_t)layer * N * F;
  const unsigned char* adjL = adjall + (size_t)layer * N * N;

  __shared__ v16bf Pbuf[8][32];                 // 8 k-tiles of 16x32 bf16, A-layout
  __shared__ unsigned char AdjT[2][16][AROW];   // double-buffered 16x256B adj tile
  __shared__ float zbuf[256];
  __shared__ float Zrow[16];

  const int myrow = r0 + li;
  const float f1i = f1[myrow];
  const float rm  = rmax[myrow];

  // staging assignment: thread t copies 16B of row (t>>4), segment (t&15)
  const int srow = threadIdx.x >> 4;
  const int sseg = (threadIdx.x & 15) * 16;
  const unsigned char* gsrc = adjL + (size_t)(r0 + srow) * N + sseg;

  const int n = 16 * w + li;                  // this wave's output column
  const __bf16* bcol = whT + (size_t)n * N;   // contiguous along j

  v8f acc = {};
  float z = 0.f;

  // prologue: stage tile 0
  copy16_to_lds(gsrc, &AdjT[0][srow][sseg]);

  const int NS = N / 256;
  for (int s = 0; s < NS; ++s) {
    const int jbase = s * 256;
    const int buf = s & 1;

    // issue next tile's async copy before waiting on the current one
    if (s + 1 < NS) {
      copy16_to_lds(gsrc + (jbase + 256), &AdjT[buf ^ 1][srow][sseg]);
      wait_async_le1();          // in-order completion => tile s has landed
    } else {
      wait_async_le0();
    }
    __syncthreads();             // tile s visible; also fences prev WMMA vs Pbuf rewrite

    // ---- produce P subtile for k-tile (s*8 + w): j0 = jbase + 32*w ----
    const int cA = 32 * w + 8 * hi;       // A elems 0..7  (col within 256-tile)
    const int cB = cA + 16;               // A elems 8..15
    uint2 ad1 = *(const uint2*)&AdjT[buf][li][cA];
    uint2 ad2 = *(const uint2*)&AdjT[buf][li][cB];
    const unsigned char* ab1 = (const unsigned char*)&ad1;
    const unsigned char* ab2 = (const unsigned char*)&ad2;
    v8f fA = *(const v8f*)(f2 + jbase + cA);
    v8f fB = *(const v8f*)(f2 + jbase + cB);
    if (s + 1 < NS) __builtin_prefetch(f2 + jbase + 256 + cA, 0, 0);

    v16bf va;
#pragma unroll
    for (int e = 0; e < 8; ++e) {
      float ev = leaky(f1i + fA[e]);
      float p = ab1[e] ? exp2f((ev - rm) * LOG2E) : 0.f;
      z += p;
      va[e] = (__bf16)p;
    }
#pragma unroll
    for (int e = 0; e < 8; ++e) {
      float ev = leaky(f1i + fB[e]);
      float p = ab2[e] ? exp2f((ev - rm) * LOG2E) : 0.f;
      z += p;
      va[8 + e] = (__bf16)p;
    }
    Pbuf[w][lane] = va;
    __syncthreads();             // Pbuf ready

    // ---- consume: 8 WMMAs over the 256-wide j super-tile ----
#pragma unroll
    for (int t = 0; t < 8; ++t) {
      v16bf A = Pbuf[t][lane];
      const int jt0 = jbase + 32 * t;
      v8bf b0 = *(const v8bf*)(bcol + jt0 + 8 * hi); v8bf b1 = *(const v8bf*)(bcol + jt0 + 16 + 8 * hi); v16bf B = __builtin_shufflevector(b0, b1, 0,1,2,3,4,5,6,7,8,9,10,11,12,13,14,15);  // two 16B halves, table order
      acc = __builtin_amdgcn_wmma_f32_16x16x32_bf16(false, A, false, B,
                                                    (short)0, acc, false, false);
    }
  }

  // ---- softmax denominator: reduce per-row Z across waves/lane-halves ----
  __syncthreads();
  zbuf[threadIdx.x] = z;
  __syncthreads();
  if (threadIdx.x < 16) {
    float Z = 0.f;
#pragma unroll
    for (int w2 = 0; w2 < 8; ++w2)
      Z += zbuf[w2 * 32 + threadIdx.x] + zbuf[w2 * 32 + 16 + threadIdx.x];
    Zrow[threadIdx.x] = Z;
  }
  __syncthreads();

#pragma unroll
  for (int v = 0; v < 8; ++v) {
    int rl = 8 * hi + v;                                 // C layout row
    float invZ = 1.0f / fmaxf(Zrow[rl], 1e-30f);
    h[(size_t)(r0 + rl) * F + n] = acc[v] * invZ;
  }
}

// ---------------------------------------------------------------------------
// Kernel 5: layernorm(q), 3 dots, 3-way softmax, mix + residual
// grid (N), block 128 (one block per output row)
// ---------------------------------------------------------------------------
__device__ __forceinline__ float blockReduceSum128(float v, float* red, int t) {
  red[t] = v;
  __syncthreads();
#pragma unroll
  for (int s = 64; s > 0; s >>= 1) {
    if (t < s) red[t] += red[t + s];
    __syncthreads();
  }
  float r = red[0];
  __syncthreads();
  return r;
}

__global__ __launch_bounds__(128) void k_final(
    const float* __restrict__ h,      // [2][N][F]
    const float* __restrict__ mpnn,   // [N][F]
    const float* __restrict__ query,  // [F]
    const float* __restrict__ gamma, const float* __restrict__ beta,
    const float* __restrict__ resp,   // [1]
    const float* __restrict__ x0,     // input_hs[0]
    float* __restrict__ out)
{
  const int i = blockIdx.x;
  const int t = threadIdx.x;
  __shared__ float red[128];

  float qv = query[t];
  float mu = blockReduceSum128(qv, red, t) * (1.0f / F);
  float d = qv - mu;
  float var = blockReduceSum128(d * d, red, t) * (1.0f / F);
  float qn = d * rsqrtf(var + 1e-5f) * gamma[t] + beta[t];

  float h0v = h[(size_t)i * F + t];
  float h1v = h[(size_t)(N + i) * F + t];
  float mv  = mpnn[(size_t)i * F + t];

  float s0 = blockReduceSum128(qn * h0v, red, t);
  float s1 = blockReduceSum128(qn * h1v, red, t);
  float s2 = blockReduceSum128(qn * mv,  red, t);

  float mx = fmaxf(s0, fmaxf(s1, s2));
  float e0 = exp2f((s0 - mx) * LOG2E);
  float e1 = exp2f((s1 - mx) * LOG2E);
  float e2 = exp2f((s2 - mx) * LOG2E);
  float inv = 1.0f / (e0 + e1 + e2);
  float mix = (e0 * h0v + e1 * h1v + e2 * mv) * inv;
  out[(size_t)i * F + t] = resp[0] * mix + x0[(size_t)i * F + t];
}

// ---------------------------------------------------------------------------
extern "C" void kernel_launch(void* const* d_in, const int* in_sizes, int n_in,
                              void* d_out, int out_size, void* d_ws, size_t ws_size,
                              hipStream_t stream) {
  (void)in_sizes; (void)n_in; (void)out_size; (void)ws_size;
  const float* input_hs        = (const float*)d_in[0];
  const unsigned char* adjs    = (const unsigned char*)d_in[1];
  const float* mpnn            = (const float*)d_in[2];
  const float* query           = (const float*)d_in[3];
  const float* W0              = (const float*)d_in[4];
  const float* a0              = (const float*)d_in[5];
  const float* W1              = (const float*)d_in[6];
  const float* a1              = (const float*)d_in[7];
  const float* gamma           = (const float*)d_in[8];
  const float* beta            = (const float*)d_in[9];
  const float* res             = (const float*)d_in[10];

  char* ws = (char*)d_ws;
  __bf16* whT = (__bf16*)ws;                               // 4 MiB
  size_t off = (size_t)2 * F * N * sizeof(__bf16);
  float* f1   = (float*)(ws + off); off += (size_t)2 * N * sizeof(float);
  float* f2   = (float*)(ws + off); off += (size_t)2 * N * sizeof(float);
  float* rmax = (float*)(ws + off); off += (size_t)2 * N * sizeof(float);
  float* h    = (float*)(ws + off);                        // 8 MiB

  k_gemm_wh<<<dim3(N / 16, 2), 256, 0, stream>>>(input_hs, W0, W1, whT);
  k_f1f2  <<<dim3(N / 256, 2), 256, 0, stream>>>(whT, a0, a1, f1, f2);
  k_rowmax<<<dim3(N, 2),       256, 0, stream>>>(adjs, f1, f2, rmax);
  k_attn  <<<dim3(N / 16, 2),  256, 0, stream>>>(adjs, whT, f1, f2, rmax, h);
  k_final <<<dim3(N),          128, 0, stream>>>(h, mpnn, query, gamma, beta,
                                                 res, input_hs, (float*)d_out);
}
